// TriangleAttention_30365418782827
// MI455X (gfx1250) — matrix operand order repaired and checked by reading, not compiled, not run on hardware
//
#include <hip/hip_runtime.h>

// ---------------------------------------------------------------------------
// TriangleAttention on gfx1250 (MI455X): BF16 WMMA pipeline, FP32 accumulate.
// N_TOK=256, C_Z=128, C=32, N_HEAD=4.
// Round-3 change: ta_gemm_rm stages the shared 128x128 BF16 weight tile into
// LDS once per workgroup via global_load_async_to_lds_b128 (ASYNCcnt +
// s_wait_asynccnt), padded to 136 elem/row for conflict-free ds_load_b128;
// WMMA B operands then come from LDS instead of 8x-amplified global reads.
// ---------------------------------------------------------------------------

typedef __attribute__((ext_vector_type(16))) __bf16 v16bf;
typedef __attribute__((ext_vector_type(8)))  __bf16 bf8v;
typedef __attribute__((ext_vector_type(4)))  __bf16 bf4v;
typedef __attribute__((ext_vector_type(8)))  float  v8f;
typedef __attribute__((ext_vector_type(4)))  float  f4v;

#define NTOK 256
#define CZ   128
#define CH   32
#define NH   4
#define NROW 65536   // NTOK*NTOK
#define BPAD 136     // padded LDS row (elements) for the staged weight tile

__device__ __forceinline__ v8f wmma_bf16(v16bf a, v16bf b, v8f c) {
  // (neg_a, A, neg_b, B, c_mod, C, reuse_a, reuse_b)
  return __builtin_amdgcn_wmma_f32_16x16x32_bf16(false, a, false, b, (short)0, c,
                                                 false, false);
}

// Load a 16-element bf16 operand register group from two 16B chunks.
__device__ __forceinline__ v16bf ld2x8(const __bf16* p_lo, const __bf16* p_hi) {
  bf8v lo = *(const bf8v*)p_lo;
  bf8v hi = *(const bf8v*)p_hi;
  v16bf r;
#pragma unroll
  for (int t = 0; t < 8; ++t) { r[t] = lo[t]; r[t + 8] = hi[t]; }
  return r;
}

// Async copy of one 16B chunk global -> LDS (CDNA5, tracked by ASYNCcnt).
__device__ __forceinline__ void async_g2l_b128(unsigned lds_byte_addr,
                                               const __bf16* gptr) {
  unsigned long long ga = (unsigned long long)(uintptr_t)gptr;
  asm volatile("global_load_async_to_lds_b128 %0, %1, off"
               :: "v"(lds_byte_addr), "v"(ga)
               : "memory");
}

__device__ __forceinline__ void wait_asynccnt0() {
  asm volatile("s_wait_asynccnt 0x0" ::: "memory");
}

// ---------------------------------------------------------------------------
// Kernel 1: transpose + bf16-convert the 5 square 128x128 weights.
// wt[mtx][n][k] = (bf16) W[mtx][k][n]   (B-operand wants column-n per lane)
// ---------------------------------------------------------------------------
__global__ void ta_prep_w(const float* __restrict__ Wq, const float* __restrict__ Wk,
                          const float* __restrict__ Wv, const float* __restrict__ Wg,
                          const float* __restrict__ Wo, __bf16* __restrict__ wt) {
  int idx = blockIdx.x * 256 + threadIdx.x;
  if (idx >= 5 * CZ * CZ) return;
  int mtx = idx >> 14;        // /16384
  int e   = idx & 16383;
  int n   = e >> 7;
  int kk  = e & 127;
  const float* W = (mtx == 0) ? Wq : (mtx == 1) ? Wk : (mtx == 2) ? Wv
                 : (mtx == 3) ? Wg : Wo;
  wt[(size_t)mtx * 16384 + (size_t)n * 128 + kk] = (__bf16)W[(size_t)kk * 128 + n];
}

// ---------------------------------------------------------------------------
// Kernel 2: LayerNorm over c_z (wave32, one wave per row) -> zn bf16,
//           plus fused b = zn @ W_b, stored transposed bws[h][row].
// ---------------------------------------------------------------------------
__global__ void ta_ln_bias(const float* __restrict__ z, const float* __restrict__ gamma,
                           const float* __restrict__ beta, const float* __restrict__ Wb,
                           __bf16* __restrict__ zn, float* __restrict__ bws) {
  int wv = threadIdx.x >> 5, lane = threadIdx.x & 31;
  long row = (long)blockIdx.x * 8 + wv;  // 0..65535
  f4v x = *(const f4v*)(z + row * CZ + lane * 4);
  float s  = x[0] + x[1] + x[2] + x[3];
  float s2 = x[0]*x[0] + x[1]*x[1] + x[2]*x[2] + x[3]*x[3];
#pragma unroll
  for (int m = 16; m >= 1; m >>= 1) {
    s  += __shfl_xor(s, m, 32);
    s2 += __shfl_xor(s2, m, 32);
  }
  float mu  = s * (1.0f / CZ);
  float var = s2 * (1.0f / CZ) - mu * mu;
  float inv = rsqrtf(var + 1e-5f);
  f4v gm = *(const f4v*)(gamma + lane * 4);
  f4v bt = *(const f4v*)(beta + lane * 4);
  const f4v* wb = (const f4v*)(Wb + lane * 16);  // 4 rows of W_b [128,4]
  float p0 = 0.f, p1 = 0.f, p2 = 0.f, p3 = 0.f;
  bf4v out;
#pragma unroll
  for (int t = 0; t < 4; ++t) {
    float v = (x[t] - mu) * inv * gm[t] + bt[t];
    out[t] = (__bf16)v;
    f4v w = wb[t];
    p0 += v * w[0]; p1 += v * w[1]; p2 += v * w[2]; p3 += v * w[3];
  }
  *(bf4v*)(zn + row * CZ + lane * 4) = out;
#pragma unroll
  for (int m = 16; m >= 1; m >>= 1) {
    p0 += __shfl_xor(p0, m, 32); p1 += __shfl_xor(p1, m, 32);
    p2 += __shfl_xor(p2, m, 32); p3 += __shfl_xor(p3, m, 32);
  }
  if (lane < 4) {
    float bv = (lane == 0) ? p0 : (lane == 1) ? p1 : (lane == 2) ? p2 : p3;
    bws[(size_t)lane * NROW + row] = bv;
  }
}

// ---------------------------------------------------------------------------
// Kernel 3: A[65536,128]bf16 @ Bt(=W^T)[128,128]bf16, per-wave 16-row strip.
// Bt is staged into LDS once per 256-thread block with async global->LDS
// copies; K=128 as 4 x v_wmma_f32_16x16x32_bf16 per 16-col tile, B operands
// double-buffered out of LDS.
// MODE 0: bf16 store (q,k)     MODE 1: transposed vT store [i][h][c][k]
// MODE 2: sigmoid -> gate bf16 MODE 3: fp32 store (final output)
// ---------------------------------------------------------------------------
template <int MODE>
__global__ void ta_gemm_rm(const __bf16* __restrict__ A, const __bf16* __restrict__ Bt,
                           void* __restrict__ outp) {
  __shared__ __bf16 bsh[CZ * BPAD];  // 128 rows x 136 elems = 34 KB
  int tid = threadIdx.x;
  int wv = tid >> 5, lane = tid & 31;
  int nl = lane & 15;
  int hiA = (lane < 16) ? 0 : 16;  // A lanes 16-31 hold K = {16..31}
  int hiB = (lane < 16) ? 0 : 16;  // B lanes 16-31 hold K = {16..31}
  int mbase = (lane < 16) ? 0 : 8; // D lanes 16-31 hold M = r+8
  long r0 = ((long)blockIdx.x * 8 + wv) * 16;

  // ---- async-stage the whole weight tile: 2048 x 16B chunks over 256 thr ----
  {
    unsigned lds_base = (unsigned)(uintptr_t)&bsh[0];
#pragma unroll
    for (int t = 0; t < 8; ++t) {
      int c = tid * 8 + t;            // chunk id 0..2047
      int n = c >> 4;                 // weight row 0..127
      int koff = (c & 15) * 8;        // element offset within row
      async_g2l_b128(lds_base + (unsigned)(n * BPAD + koff) * 2,
                     Bt + (size_t)n * CZ + koff);
    }
  }

  // A operands from global overlap the async DMA
  v16bf a[4];
#pragma unroll
  for (int kc = 0; kc < 4; ++kc) {
    const __bf16* p = A + (r0 + nl) * CZ + kc * 32 + hiA;
    a[kc] = ld2x8(p, p + 8);
  }

  wait_asynccnt0();
  __syncthreads();

  const __bf16* brow = &bsh[(size_t)nl * BPAD + hiB];  // lane's B row base (LDS)
#pragma unroll
  for (int nt = 0; nt < 8; ++nt) {
    const __bf16* bp = brow + (size_t)nt * 16 * BPAD;
    v8f acc = {};
    v16bf bcur = ld2x8(bp, bp + 8);                  // kc = 0
#pragma unroll
    for (int kc = 0; kc < 4; ++kc) {
      v16bf bnext;
      if (kc < 3) {                                  // prefetch kc+1 before WMMA
        const __bf16* pn = bp + (kc + 1) * 32;
        bnext = ld2x8(pn, pn + 8);
      }
      acc = wmma_bf16(a[kc], bcur, acc);
      if (kc < 3) bcur = bnext;
    }
#pragma unroll
    for (int r = 0; r < 8; ++r) {
      int m = r + mbase;
      long row = r0 + m;
      int col = nt * 16 + nl;
      float v = acc[r];
      if (MODE == 0) {
        ((__bf16*)outp)[row * CZ + col] = (__bf16)v;
      } else if (MODE == 1) {
        long i = row >> 8; int j = (int)(row & 255);
        int h = col >> 5, c = col & 31;
        ((__bf16*)outp)[(((i * NH + h) * CH + c) << 8) + j] = (__bf16)v;
      } else if (MODE == 2) {
        float g = 1.0f / (1.0f + __expf(-v));
        ((__bf16*)outp)[row * CZ + col] = (__bf16)g;
      } else {
        ((float*)outp)[row * CZ + col] = v;
      }
    }
  }
}

// ---------------------------------------------------------------------------
// Kernel 4: attention per (i, h, 32-row j-block). 2 waves of 16 j-rows each.
// S = Q K^T / sqrt(C) + b[j,k,h] + mask   (16 WMMA, K-tiles double-buffered)
// softmax over k (2 lanes per row) in LDS, then O = P V (16 WMMA, A from LDS
// via ds_load_b128, V tiles double-buffered), gated epilogue.
// ---------------------------------------------------------------------------
__global__ void ta_attn(const __bf16* __restrict__ q, const __bf16* __restrict__ k,
                        const __bf16* __restrict__ vT, const __bf16* __restrict__ gate,
                        const float* __restrict__ bws, const float* __restrict__ mask,
                        __bf16* __restrict__ att) {
  __shared__ float  sc[2 * 16 * NTOK];   // 32 KB scores
  __shared__ __bf16 pr[2 * 16 * NTOK];   // 16 KB probs
  int wv = threadIdx.x >> 5, lane = threadIdx.x & 31;
  int i = blockIdx.x, h = blockIdx.y, jb = blockIdx.z;
  int j0 = jb * 32 + wv * 16;
  int nl = lane & 15;
  int hiA = (lane < 16) ? 0 : 16;
  int hiB = (lane < 16) ? 0 : 16;
  int mbase = (lane < 16) ? 0 : 8;
  const float scl = 0.17677669529663687f;  // 1/sqrt(32)

  const __bf16* qp = q + ((long)(i * NTOK + j0 + nl)) * CZ + h * CH + hiA;
  v16bf aq = ld2x8(qp, qp + 8);

  // ---- S = Q K^T, K-tiles double-buffered ----
  const __bf16* krow = k + ((long)(i * NTOK + nl)) * CZ + h * CH + hiB;
  v16bf bk = ld2x8(krow, krow + 8);  // kt = 0
#pragma unroll 4
  for (int kt = 0; kt < 16; ++kt) {
    v16bf bnext;
    if (kt < 15) {
      const __bf16* kp = krow + (size_t)(kt + 1) * 16 * CZ;
      bnext = ld2x8(kp, kp + 8);
    }
    v8f acc = {};
    acc = wmma_bf16(aq, bk, acc);
    if (kt < 15) bk = bnext;
    int kk = kt * 16 + nl;
    float mterm = -1e9f * (1.0f - mask[kk]);
    const float* brow = bws + (size_t)h * NROW + kk;
#pragma unroll
    for (int r = 0; r < 8; ++r) {
      int m = r + mbase;
      int j = j0 + m;
      float s = acc[r] * scl + brow[(size_t)j * NTOK] + mterm;
      sc[(wv * 16 + m) * NTOK + kk] = s;
    }
  }
  __syncthreads();

  {  // softmax over k=256: 2 lanes per row
    int row = lane >> 1, half = lane & 1;
    float* sr = &sc[(wv * 16 + row) * NTOK + half * 128];
    float mx = -3.4e38f;
    for (int t = 0; t < 128; ++t) mx = fmaxf(mx, sr[t]);
    mx = fmaxf(mx, __shfl_xor(mx, 1, 32));
    float sum = 0.f;
    for (int t = 0; t < 128; ++t) sum += __expf(sr[t] - mx);
    sum += __shfl_xor(sum, 1, 32);
    float invs = 1.0f / sum;
    __bf16* pw = &pr[(wv * 16 + row) * NTOK + half * 128];
    for (int t = 0; t < 128; ++t) pw[t] = (__bf16)(__expf(sr[t] - mx) * invs);
  }
  __syncthreads();

  // ---- O = P V, A from LDS, V tiles double-buffered ----
  v8f o0 = {}, o1 = {};
  const __bf16* vbase = vT + ((long)(i * NH + h)) * CH * NTOK;
  const __bf16* v0row = vbase + (size_t)nl * NTOK + hiB;
  const __bf16* v1row = vbase + (size_t)(16 + nl) * NTOK + hiB;
  v16bf bv0 = ld2x8(v0row, v0row + 8);
  v16bf bv1 = ld2x8(v1row, v1row + 8);
#pragma unroll 4
  for (int kc = 0; kc < 8; ++kc) {
    const __bf16* ap = &pr[(wv * 16 + nl) * NTOK + kc * 32 + hiA];
    v16bf a16 = ld2x8(ap, ap + 8);
    v16bf n0, n1;
    if (kc < 7) {
      const __bf16* p0 = v0row + (kc + 1) * 32;
      const __bf16* p1 = v1row + (kc + 1) * 32;
      n0 = ld2x8(p0, p0 + 8);
      n1 = ld2x8(p1, p1 + 8);
    }
    o0 = wmma_bf16(a16, bv0, o0);
    o1 = wmma_bf16(a16, bv1, o1);
    if (kc < 7) { bv0 = n0; bv1 = n1; }
  }

#pragma unroll
  for (int r = 0; r < 8; ++r) {
    int m = r + mbase;
    long j = j0 + m;
    long base = ((long)i * NTOK + j) * CZ + h * CH;
    float g0 = (float)gate[base + nl];
    float g1 = (float)gate[base + 16 + nl];
    att[base + nl]      = (__bf16)(o0[r] * g0);
    att[base + 16 + nl] = (__bf16)(o1[r] * g1);
  }
}

// ---------------------------------------------------------------------------
// Host launcher
// ---------------------------------------------------------------------------
extern "C" void kernel_launch(void* const* d_in, const int* in_sizes, int n_in,
                              void* d_out, int out_size, void* d_ws, size_t ws_size,
                              hipStream_t stream) {
  (void)in_sizes; (void)n_in; (void)out_size; (void)ws_size;
  const float* z     = (const float*)d_in[0];
  const float* smask = (const float*)d_in[1];
  const float* gamma = (const float*)d_in[2];
  const float* beta  = (const float*)d_in[3];
  const float* Wq    = (const float*)d_in[4];
  const float* Wk    = (const float*)d_in[5];
  const float* Wv    = (const float*)d_in[6];
  const float* Wb    = (const float*)d_in[7];
  const float* Wg    = (const float*)d_in[8];
  const float* Wo    = (const float*)d_in[9];

  char* ws = (char*)d_ws;
  const size_t MB = 1024 * 1024;
  __bf16* zn  = (__bf16*)(ws + 0 * MB);    // 16 MB
  __bf16* qw  = (__bf16*)(ws + 16 * MB);   // 16 MB
  __bf16* kw  = (__bf16*)(ws + 32 * MB);   // 16 MB
  __bf16* vT  = (__bf16*)(ws + 48 * MB);   // 16 MB  [i][h][c][k]
  __bf16* gt  = (__bf16*)(ws + 64 * MB);   // 16 MB  sigmoid(g)
  __bf16* att = (__bf16*)(ws + 80 * MB);   // 16 MB  gated attention output
  float*  bws = (float*)(ws + 96 * MB);    //  1 MB  bias [h][65536]
  __bf16* wt  = (__bf16*)(ws + 98 * MB);   // 160 KB transposed bf16 weights

  ta_prep_w<<<(5 * 16384 + 255) / 256, 256, 0, stream>>>(Wq, Wk, Wv, Wg, Wo, wt);
  ta_ln_bias<<<NROW / 8, 256, 0, stream>>>(z, gamma, beta, Wb, zn, bws);
  ta_gemm_rm<0><<<512, 256, 0, stream>>>(zn, wt + 0 * 16384, qw);
  ta_gemm_rm<0><<<512, 256, 0, stream>>>(zn, wt + 1 * 16384, kw);
  ta_gemm_rm<1><<<512, 256, 0, stream>>>(zn, wt + 2 * 16384, vT);
  ta_gemm_rm<2><<<512, 256, 0, stream>>>(zn, wt + 3 * 16384, gt);
  ta_attn<<<dim3(NTOK, NH, 8), 64, 0, stream>>>(qw, kw, vT, gt, bws, smask, att);
  ta_gemm_rm<3><<<512, 256, 0, stream>>>(att, wt + 4 * 16384, d_out);
}
